// LSTM_76381698392570
// MI455X (gfx1250) — matrix operand order repaired and checked by reading, not compiled, not run on hardware
//
#include <hip/hip_runtime.h>
#include <math.h>

typedef __attribute__((ext_vector_type(16))) __bf16        v16bf;
typedef __attribute__((ext_vector_type(8)))  float         v8f;
typedef __attribute__((ext_vector_type(4)))  unsigned int  v4u;

namespace {
constexpr int Bx = 128, Sx = 256, Dx = 512, Hx = 1024, NCx = 128;
constexpr int KB_H   = Hx / 32;        // 32 k-blocks from hidden state
constexpr int KB_X   = Dx / 32;        // 16 k-blocks from embedding
constexpr int KB_TOT = KB_H + KB_X;    // 48
constexpr int KB_P   = Hx / 32;        // 32 k-blocks for projection
constexpr int NB_P   = NCx / 16;       // 8 n-blocks for projection
}

union Frag { v16bf v; v4u q[2]; };

#if __has_builtin(__builtin_amdgcn_global_load_async_to_lds_b128)
#define ASYNC_LDS 1
typedef __attribute__((ext_vector_type(4))) int v4i;
typedef __attribute__((address_space(3))) v4i lds_v4i;
typedef __attribute__((address_space(1))) v4i glob_v4i;
#define ASYNC_CP(gp, lp) \
  __builtin_amdgcn_global_load_async_to_lds_b128((glob_v4i*)(void*)(gp), \
                                                 (lds_v4i*)(void*)(lp), 0, 0)
#else
#define ASYNC_LDS 0
#endif

__device__ __forceinline__ void wait_async0() {
#if __has_builtin(__builtin_amdgcn_s_wait_asynccnt)
  __builtin_amdgcn_s_wait_asynccnt(0);
#else
  asm volatile("s_wait_asynccnt 0x0" ::: "memory");
#endif
}

__device__ __forceinline__ unsigned short f2bf(float f) {
  union { float f; unsigned int u; } v; v.f = f;
  unsigned int r = v.u + 0x7FFFu + ((v.u >> 16) & 1u);   // round-to-nearest-even
  return (unsigned short)(r >> 16);
}
__device__ __forceinline__ float sigm(float x) { return 1.0f / (1.0f + expf(-x)); }

// ---- zero h0 and C ---------------------------------------------------------
__global__ void k_init(unsigned short* __restrict__ h0, float* __restrict__ C) {
  int t = blockIdx.x * blockDim.x + threadIdx.x;
  if (t < Bx * Hx) { h0[t] = 0; C[t] = 0.0f; }
}

// ---- embedding gather -> bf16 [B,S,D] -------------------------------------
__global__ void k_gather_embed(const int* __restrict__ x, const float* __restrict__ emb,
                               unsigned short* __restrict__ xe) {
  long t = (long)blockIdx.x * blockDim.x + threadIdx.x;  // B*S*D threads
  int  d  = (int)(t % Dx);
  long bs = t / Dx;                                      // b*S + s
  int  idx = x[bs];
  xe[t] = f2bf(emb[(long)idx * Dx + d]);
}

// ---- pack [Wh;Wx] for 4 gates into WMMA B fragments (bf16) -----------------
// Layout: [hT 0..63][kb 0..47][g 0..3][lane 0..31][16 halves]
// N=lane%16; K = 8*(lane/16) + j for j<8, 16 + 8*(lane/16) + (j-8) for j>=8  (ISA B layout, K=32)
__global__ void k_pack_gates(const float* __restrict__ Wfx, const float* __restrict__ Wfh,
                             const float* __restrict__ Wix, const float* __restrict__ Wih,
                             const float* __restrict__ Wgx, const float* __restrict__ Wgh,
                             const float* __restrict__ Wox, const float* __restrict__ Woh,
                             unsigned short* __restrict__ Wfrag) {
  long t = (long)blockIdx.x * blockDim.x + threadIdx.x;  // 64*48*4*32*16
  int j    = (int)(t & 15);
  int lane = (int)((t >> 4) & 31);
  int g    = (int)((t >> 9) & 3);
  long fb  = t >> 11;
  int kb   = (int)(fb % KB_TOT);
  int hT   = (int)(fb / KB_TOT);
  int col  = (hT << 4) + (lane & 15);
  int k    = kb * 32 + ((lane >= 16) ? 8 : 0) + ((j < 8) ? j : j + 8);
  const float* Wh; const float* Wx;
  if      (g == 0) { Wh = Wfh; Wx = Wfx; }
  else if (g == 1) { Wh = Wih; Wx = Wix; }
  else if (g == 2) { Wh = Wgh; Wx = Wgx; }
  else             { Wh = Woh; Wx = Wox; }
  float v = (k < Hx) ? Wh[(long)k * Hx + col] : Wx[(long)(k - Hx) * Hx + col];
  Wfrag[t] = f2bf(v);
}

// ---- pack Wph [H,NC] into WMMA B fragments (layout [nb][kb][lane][16]) -----
__global__ void k_pack_ph(const float* __restrict__ Wph, unsigned short* __restrict__ Wpf) {
  long t = (long)blockIdx.x * blockDim.x + threadIdx.x;  // NB_P*KB_P*512
  int j    = (int)(t & 15);
  int lane = (int)((t >> 4) & 31);
  long fb  = t >> 9;
  int kb   = (int)(fb % KB_P);
  int nb   = (int)(fb / KB_P);
  int col  = (nb << 4) + (lane & 15);
  int k    = kb * 32 + ((lane >= 16) ? 8 : 0) + ((j < 8) ? j : j + 8);
  Wpf[t] = f2bf(Wph[(long)k * NCx + col]);
}

// ---- one LSTM timestep -----------------------------------------------------
// grid: 32 blocks x 512 thr = 16 waves/block; wave = (hL 0..1, mT 0..7)
// Block covers hT = blockIdx*2 + hL; weights + A slice staged via LDS.
__global__ void __launch_bounds__(512)
k_lstm_step(const unsigned short* __restrict__ Wfrag,
            const unsigned short* __restrict__ xe,
            const unsigned short* __restrict__ h_in,
            unsigned short* __restrict__ h_out,
            float* __restrict__ Cst,
            const float* __restrict__ bfv, const float* __restrict__ biv,
            const float* __restrict__ bgv, const float* __restrict__ bov,
            const int* __restrict__ xin, int s) {
  __shared__ v4u ldsB[2][512];   // per buf: 2 hT x 4 gates x 32 lanes x 16 halves = 8KB
  __shared__ v4u ldsA[2][512];   // per buf: 128 rows x 32 halves (one k-block)  = 8KB

  const int t    = threadIdx.x;         // 0..511
  const int lane = t & 31;
  const int wave = t >> 5;              // 0..15
  const int mT   = wave & 7;            // batch tile
  const int hL   = wave >> 3;           // 0..1
  const int hT   = (blockIdx.x << 1) + hL;

  // staging decode (per thread): B = one v4u of block's 8KB slice; A = row/part
  const int aRow  = t >> 2;             // 0..127
  const int aPart = t & 3;              // 16B part within 64B row-slice
  const v4u* bbase = (const v4u*)Wfrag +
                     ((long)((blockIdx.x << 1) + (t >> 8)) * KB_TOT) * 256 + (t & 255);

  auto a_src = [&](int kb) -> const v4u* {
    if (kb < KB_H) return (const v4u*)h_in + (long)aRow * (Hx / 8) + kb * 4 + aPart;
    return (const v4u*)xe + ((long)aRow * Sx + s) * (Dx / 8) + (kb - KB_H) * 4 + aPart;
  };

  // fragment read offsets
  const int aOff = (((mT << 4) + (lane & 15)) << 5) + ((lane >= 16) ? 8 : 0); // halves
  const int bOff = (hL << 8) + (lane << 1);                                   // v4u units

  // ---- prologue: stage kb=0 into buffer 0 ----
#if ASYNC_LDS
  ASYNC_CP(bbase,    &ldsB[0][t]);
  ASYNC_CP(a_src(0), &ldsA[0][t]);
  wait_async0();
#else
  ldsB[0][t] = *bbase;
  ldsA[0][t] = *a_src(0);
#endif
  __syncthreads();

  v8f acc[4] = {};
  for (int kb = 0; kb < KB_TOT; ++kb) {
    const int cur = kb & 1;
    const int nxt = cur ^ 1;
    const bool have_next = (kb + 1 < KB_TOT);
#if !ASYNC_LDS
    v4u ta, tb;
#endif
    if (have_next) {
      const v4u* gb = bbase + (kb + 1) * 256;
      const v4u* ga = a_src(kb + 1);
#if ASYNC_LDS
      ASYNC_CP(gb, &ldsB[nxt][t]);
      ASYNC_CP(ga, &ldsA[nxt][t]);
#else
      tb = *gb; ta = *ga;
#endif
    }
    // consume current buffer
    Frag A;
    const unsigned short* la = (const unsigned short*)&ldsA[cur][0] + aOff;
    A.q[0] = *(const v4u*)la;
    A.q[1] = *(const v4u*)(la + 16);
#pragma unroll
    for (int g = 0; g < 4; ++g) {
      const v4u* bq = &ldsB[cur][bOff + (g << 6)];
      Frag Bv; Bv.q[0] = bq[0]; Bv.q[1] = bq[1];
      acc[g] = __builtin_amdgcn_wmma_f32_16x16x32_bf16(false, A.v, false, Bv.v,
                                                       (short)0, acc[g], false, false);
    }
#if ASYNC_LDS
    if (have_next) wait_async0();
#else
    if (have_next) { ldsB[nxt][t] = tb; ldsA[nxt][t] = ta; }
#endif
    __syncthreads();
  }

  // Pointwise LSTM update; each lane owns 8 (b, hc) cells (ISA C/D layout)
  const int   hc  = (hT << 4) + (lane & 15);
  const float bfs = bfv[hc], bis = biv[hc], bgs = bgv[hc], bos = bov[hc];
  const int   rbase = (mT << 4) + ((lane >= 16) ? 8 : 0);
#pragma unroll
  for (int j = 0; j < 8; ++j) {
    const int r  = rbase + j;
    float f  = sigm(acc[0][j] + bfs);
    float i  = sigm(acc[1][j] + bis);
    float g2 = sigm(acc[2][j] + bgs);   // reference uses sigmoid here, not tanh
    float o  = sigm(acc[3][j] + bos);
    float rst = (xin[(long)r * Sx + s] > 0) ? 1.0f : 0.0f;
    long ci = (long)r * Hx + hc;
    float Cn = (g2 * i + Cst[ci] * f) * rst;
    Cst[ci]   = Cn;
    h_out[ci] = f2bf(o * tanhf(Cn));
  }
}

// ---- head: P = h_final @ Wph + bp  (WMMA) ----------------------------------
__global__ void k_proj(const unsigned short* __restrict__ hfin,
                       const unsigned short* __restrict__ Wpf,
                       const float* __restrict__ bp, float* __restrict__ P) {
  const int wave  = (int)((blockIdx.x * blockDim.x + threadIdx.x) >> 5); // 0..63
  const int lane  = threadIdx.x & 31;
  const int mT    = wave >> 3;
  const int nT    = wave & 7;
  const int mRowA = (mT << 4) + (lane & 15);
  const int klo   = (lane >= 16) ? 8 : 0;
  v8f acc = {};
  const unsigned short* hrow = hfin + (long)mRowA * Hx;
  for (int kb = 0; kb < KB_P; ++kb) {
    Frag A;
    const unsigned short* ap = hrow + kb * 32 + klo;
    A.q[0] = *(const v4u*)(ap);
    A.q[1] = *(const v4u*)(ap + 16);
    Frag Bv;
    const unsigned short* wp = Wpf + (((long)nT * KB_P + kb) * 32 + lane) * 16;
    Bv.q[0] = *(const v4u*)(wp);
    Bv.q[1] = *(const v4u*)(wp + 8);
    acc = __builtin_amdgcn_wmma_f32_16x16x32_bf16(false, A.v, false, Bv.v,
                                                  (short)0, acc, false, false);
  }
  const int c     = (nT << 4) + (lane & 15);
  const int rbase = (mT << 4) + ((lane >= 16) ? 8 : 0);
  const float bc  = bp[c];
#pragma unroll
  for (int j = 0; j < 8; ++j) P[(long)(rbase + j) * NCx + c] = acc[j] + bc;
}

// ---- log_softmax over NC=128, one wave per row -----------------------------
__global__ void k_logsoftmax(const float* __restrict__ P, float* __restrict__ out) {
  const int wave = (int)((blockIdx.x * blockDim.x + threadIdx.x) >> 5);
  const int lane = threadIdx.x & 31;
  if (wave >= Bx) return;
  const float* row = P + (long)wave * NCx;
  float v[4];
  float mx = -INFINITY;
#pragma unroll
  for (int t = 0; t < 4; ++t) { v[t] = row[lane + 32 * t]; mx = fmaxf(mx, v[t]); }
  for (int off = 16; off >= 1; off >>= 1) mx = fmaxf(mx, __shfl_xor(mx, off, 32));
  float sum = 0.0f;
#pragma unroll
  for (int t = 0; t < 4; ++t) sum += expf(v[t] - mx);
  for (int off = 16; off >= 1; off >>= 1) sum += __shfl_xor(sum, off, 32);
  float ls = logf(sum);
#pragma unroll
  for (int t = 0; t < 4; ++t) out[(long)wave * NCx + lane + 32 * t] = v[t] - mx - ls;
}

extern "C" void kernel_launch(void* const* d_in, const int* in_sizes, int n_in,
                              void* d_out, int out_size, void* d_ws, size_t ws_size,
                              hipStream_t stream) {
  (void)in_sizes; (void)n_in; (void)out_size; (void)ws_size;
  const int*   x   = (const int*)  d_in[0];
  const float* emb = (const float*)d_in[1];
  const float* Wfx = (const float*)d_in[2];
  const float* Wfh = (const float*)d_in[3];
  const float* bf  = (const float*)d_in[4];
  const float* Wix = (const float*)d_in[5];
  const float* Wih = (const float*)d_in[6];
  const float* bi  = (const float*)d_in[7];
  const float* Wgx = (const float*)d_in[8];
  const float* Wgh = (const float*)d_in[9];
  const float* bg  = (const float*)d_in[10];
  const float* Wox = (const float*)d_in[11];
  const float* Woh = (const float*)d_in[12];
  const float* bo  = (const float*)d_in[13];
  const float* Wph = (const float*)d_in[14];
  const float* bp  = (const float*)d_in[15];

  char* ws = (char*)d_ws;
  size_t off = 0;
  auto carve = [&](size_t bytes) -> void* {
    void* p = ws + off;
    off = (off + bytes + 255) & ~(size_t)255;
    return p;
  };
  unsigned short* xe    = (unsigned short*)carve((size_t)Bx * Sx * Dx * 2);          // 33.5 MB
  unsigned short* Wfrag = (unsigned short*)carve((size_t)64 * KB_TOT * 4 * 512 * 2); // 12.6 MB
  unsigned short* Wpf   = (unsigned short*)carve((size_t)NB_P * KB_P * 512 * 2);
  unsigned short* h0    = (unsigned short*)carve((size_t)Bx * Hx * 2);
  unsigned short* h1    = (unsigned short*)carve((size_t)Bx * Hx * 2);
  float*          Cst   = (float*)carve((size_t)Bx * Hx * 4);
  float*          P     = (float*)carve((size_t)Bx * NCx * 4);

  k_init<<<(Bx * Hx + 255) / 256, 256, 0, stream>>>(h0, Cst);
  k_gather_embed<<<((long)Bx * Sx * Dx) / 256, 256, 0, stream>>>(x, emb, xe);
  k_pack_gates<<<((long)64 * KB_TOT * 4 * 512) / 256, 256, 0, stream>>>(
      Wfx, Wfh, Wix, Wih, Wgx, Wgh, Wox, Woh, Wfrag);
  k_pack_ph<<<((long)NB_P * KB_P * 512) / 256, 256, 0, stream>>>(Wph, Wpf);

  for (int s = 0; s < Sx; ++s) {
    const unsigned short* hin  = (s & 1) ? h1 : h0;
    unsigned short*       hout = (s & 1) ? h0 : h1;
    k_lstm_step<<<32, 512, 0, stream>>>(Wfrag, xe, hin, hout, Cst,
                                        bf, bi, bg, bo, x, s);
  }
  // S=256 is even -> final h lives in h0
  k_proj<<<8, 256, 0, stream>>>(h0, Wpf, bp, P);
  k_logsoftmax<<<16, 256, 0, stream>>>(P, (float*)d_out);
}
